// MultiHeadSelfAttention_9010841387449
// MI455X (gfx1250) — matrix operand order repaired and checked by reading, not compiled, not run on hardware
//
#include <hip/hip_runtime.h>
#include <hip/hip_bf16.h>

// MHSA block for MI455X (gfx1250, wave32).
// All GEMM stages use v_wmma_f32_16x16x32_f16 (f16 in, f32 accumulate).
// Softmax / residual / LayerNorm in f32.

#define D_MODEL 1024
#define S_LEN   2048
#define NBATCH  2
#define NHEADS  16
#define DHEAD   64
#define NTOK    (NBATCH * S_LEN)   // 4096

typedef __attribute__((ext_vector_type(8)))  _Float16 v8h;
typedef __attribute__((ext_vector_type(16))) _Float16 v16h;
typedef __attribute__((ext_vector_type(8)))  float    v8f;

static __device__ __forceinline__ v8f vzero8() {
  v8f z;
#pragma unroll
  for (int i = 0; i < 8; ++i) z[i] = 0.0f;
  return z;
}

// Load 16 halves as two 8-half chunks at p and p+gap.
// A-operand layout (16x32 f16): elements 0..7 at base, 8..15 at base+16 (gap=16).
// B-operand layout (32x16 f16): same split as A (gap=16).
static __device__ __forceinline__ v16h ld_split16(const _Float16* p, int gap) {
  v8h lo = *(const v8h*)(p);
  v8h hi = *(const v8h*)(p + gap);
  return __builtin_shufflevector(lo, hi, 0,1,2,3,4,5,6,7,8,9,10,11,12,13,14,15);
}

static __device__ __forceinline__ v8f wmma16(v16h a, v16h b, v8f c) {
  // (neg_a, A, neg_b, B, c_mod, C, reuse_a, reuse_b)
  return __builtin_amdgcn_wmma_f32_16x16x32_f16(false, a, false, b, (short)0, c,
                                                false, false);
}

// ---------------------------------------------------------------- pack f32->f16
__global__ void __launch_bounds__(256)
pack_f16_kernel(const float* __restrict__ in, _Float16* __restrict__ out, int n) {
  int i = blockIdx.x * 256 + threadIdx.x;
  if (i < n) out[i] = (_Float16)in[i];
}

// ---------------------------------------------------------------- WMMA GEMM
// C[m, n] = sum_k A[m, k] * W[n, k] + bias[n]   (torch Linear: x @ W^T + b)
// One wave computes a 32x64 tile. K = D_MODEL = 1024.
// mode 0: Q  -> f16 [b,h,s,dh], scaled by 0.125 (1/sqrt(Dh) folded in)
// mode 1: K  -> f16 [b,h,s,dh]
// mode 2: V  -> f16 [b,h,dh,s]  (transposed for contiguous PV B-operand loads)
// mode 3: out-proj -> f32 [tok, e] with +bias +residual
__global__ void __launch_bounds__(32)
gemm_xw_kernel(const _Float16* __restrict__ A, const _Float16* __restrict__ W,
               const float* __restrict__ bias, const float* __restrict__ resid,
               _Float16* __restrict__ out_h, float* __restrict__ out_f, int mode) {
  const int lane = threadIdx.x & 31;
  const int hl   = lane >> 4;
  const int l16  = lane & 15;
  const int m0   = blockIdx.x * 32;
  const int n0   = blockIdx.y * 64;

  const int aoff = hl ? 8 : 0;    // A operand K-offset for this lane half
  const int boff = hl ? 8 : 0;    // B operand K-offset for this lane half

  v8f acc[2][4];
#pragma unroll
  for (int mt = 0; mt < 2; ++mt)
#pragma unroll
    for (int nt = 0; nt < 4; ++nt) acc[mt][nt] = vzero8();

  for (int kc = 0; kc < D_MODEL; kc += 32) {
    v16h a0 = ld_split16(A + (size_t)(m0 + l16)      * D_MODEL + kc + aoff, 16);
    v16h a1 = ld_split16(A + (size_t)(m0 + 16 + l16) * D_MODEL + kc + aoff, 16);
#pragma unroll
    for (int nt = 0; nt < 4; ++nt) {
      v16h b = ld_split16(W + (size_t)(n0 + nt * 16 + l16) * D_MODEL + kc + boff, 16);
      acc[0][nt] = wmma16(a0, b, acc[0][nt]);
      acc[1][nt] = wmma16(a1, b, acc[1][nt]);
    }
  }

#pragma unroll
  for (int mt = 0; mt < 2; ++mt) {
#pragma unroll
    for (int nt = 0; nt < 4; ++nt) {
#pragma unroll
      for (int r = 0; r < 8; ++r) {
        const int m = m0 + 16 * mt + r + 8 * hl;   // token index
        const int n = n0 + 16 * nt + l16;          // output feature e
        float v = acc[mt][nt][r] + bias[n];
        if (mode == 3) {
          out_f[(size_t)m * D_MODEL + n] = v + resid[(size_t)m * D_MODEL + n];
        } else {
          const int b = m >> 11, s = m & 2047;
          const int h = n >> 6,  dh = n & 63;
          if (mode == 0) {
            out_h[(((size_t)(b * NHEADS + h) * S_LEN) + s) * DHEAD + dh] =
                (_Float16)(v * 0.125f);
          } else if (mode == 1) {
            out_h[(((size_t)(b * NHEADS + h) * S_LEN) + s) * DHEAD + dh] = (_Float16)v;
          } else {
            out_h[(((size_t)(b * NHEADS + h) * DHEAD) + dh) * S_LEN + s] = (_Float16)v;
          }
        }
      }
    }
  }
}

// ---------------------------------------------------------------- flash attention
// One wave per (b, h, 16-query tile). Online softmax over 32-key chunks.
__global__ void __launch_bounds__(32)
attn_kernel(const _Float16* __restrict__ Qh, const _Float16* __restrict__ Kh,
            const _Float16* __restrict__ Vt, _Float16* __restrict__ attnh) {
  const int lane = threadIdx.x & 31;
  const int hl   = lane >> 4;
  const int l16  = lane & 15;
  const int q0   = blockIdx.x * 16;
  const int h    = blockIdx.y;
  const int b    = blockIdx.z;
  const size_t hb = (size_t)(b * NHEADS + h) * S_LEN;   // row base in Qh/Kh
  const size_t vb = (size_t)(b * NHEADS + h) * DHEAD;   // row base in Vt

  __shared__ __align__(16) _Float16 Pl[16 * 32];

  const int aoff = hl ? 8 : 0;
  const int boff = hl ? 8 : 0;

  // Q tile 16x64 as two A operands (kd 0..31, 32..63); scale folded in earlier.
  const _Float16* qrow = Qh + (hb + q0 + l16) * DHEAD;
  v16h aq0 = ld_split16(qrow + aoff, 16);
  v16h aq1 = ld_split16(qrow + 32 + aoff, 16);

  v8f accO[4];
  float mrun[8], lrun[8];
#pragma unroll
  for (int nt = 0; nt < 4; ++nt) accO[nt] = vzero8();
#pragma unroll
  for (int r = 0; r < 8; ++r) { mrun[r] = -1e30f; lrun[r] = 0.0f; }

  for (int kv = 0; kv < S_LEN; kv += 32) {
    // scores: two 16x16 key subtiles, K-dim 64 via chained WMMA
    v8f sc[2];
#pragma unroll
    for (int c = 0; c < 2; ++c) {
      const _Float16* krow = Kh + (hb + kv + 16 * c + l16) * DHEAD;
      v16h bk0 = ld_split16(krow + boff, 16);        // kd 0..31
      v16h bk1 = ld_split16(krow + 32 + boff, 16);   // kd 32..63
      sc[c] = wmma16(aq1, bk1, wmma16(aq0, bk0, vzero8()));
    }

    // online softmax per row r (row = r + 8*hl); cols live across 16-lane half
#pragma unroll
    for (int r = 0; r < 8; ++r) {
      float s0 = sc[0][r], s1 = sc[1][r];
      float mx = fmaxf(s0, s1);
#pragma unroll
      for (int off = 8; off >= 1; off >>= 1)
        mx = fmaxf(mx, __shfl_xor(mx, off, 32));
      float nm   = fmaxf(mrun[r], mx);
      float corr = __expf(mrun[r] - nm);
      mrun[r] = nm;
      float p0 = __expf(s0 - nm), p1 = __expf(s1 - nm);
      float rs = p0 + p1;
#pragma unroll
      for (int off = 8; off >= 1; off >>= 1) rs += __shfl_xor(rs, off, 32);
      lrun[r] = lrun[r] * corr + rs;
#pragma unroll
      for (int nt = 0; nt < 4; ++nt) accO[nt][r] *= corr;
      const int row = r + 8 * hl;
      Pl[row * 32 + l16]      = (_Float16)p0;
      Pl[row * 32 + 16 + l16] = (_Float16)p1;
    }
    __syncthreads();

    // reload P in A layout, then P[16x32] @ V[32x64]
    v16h ap = ld_split16(&Pl[l16 * 32 + aoff], 16);
#pragma unroll
    for (int nt = 0; nt < 4; ++nt) {
      const _Float16* vrow = Vt + (vb + nt * 16 + l16) * S_LEN + kv;
      v16h bv = ld_split16(vrow + boff, 16);
      accO[nt] = wmma16(ap, bv, accO[nt]);
    }
    __syncthreads();
  }

  // normalize and write attn output in [B,S,D] f16 (A-ready for final GEMM)
#pragma unroll
  for (int nt = 0; nt < 4; ++nt) {
#pragma unroll
    for (int r = 0; r < 8; ++r) {
      const int s = q0 + r + 8 * hl;
      const int t = b * S_LEN + s;
      const int e = h * DHEAD + nt * 16 + l16;
      attnh[(size_t)t * D_MODEL + e] = (_Float16)(accO[nt][r] / lrun[r]);
    }
  }
}

// ---------------------------------------------------------------- LayerNorm
__global__ void __launch_bounds__(256)
ln_kernel(const float* __restrict__ y, const float* __restrict__ gamma,
          const float* __restrict__ beta, float* __restrict__ out) {
  __shared__ float sS[8], sS2[8];
  const int t = blockIdx.x;
  const float* row = y + (size_t)t * D_MODEL;
  float s = 0.0f, s2 = 0.0f;
  for (int i = threadIdx.x; i < D_MODEL; i += 256) {
    float v = row[i];
    s += v; s2 += v * v;
  }
#pragma unroll
  for (int off = 16; off >= 1; off >>= 1) {
    s  += __shfl_xor(s,  off, 32);
    s2 += __shfl_xor(s2, off, 32);
  }
  const int w = threadIdx.x >> 5;
  if ((threadIdx.x & 31) == 0) { sS[w] = s; sS2[w] = s2; }
  __syncthreads();
  if (w == 0) {
    s  = sS[threadIdx.x & 7];
    s2 = sS2[threadIdx.x & 7];
#pragma unroll
    for (int off = 4; off >= 1; off >>= 1) {
      s  += __shfl_xor(s,  off, 32);
      s2 += __shfl_xor(s2, off, 32);
    }
    if (threadIdx.x == 0) { sS[0] = s; sS2[0] = s2; }
  }
  __syncthreads();
  s = sS[0]; s2 = sS2[0];
  const float mu  = s  * (1.0f / D_MODEL);
  const float var = s2 * (1.0f / D_MODEL) - mu * mu;
  const float rs  = rsqrtf(var + 1e-5f);
  for (int i = threadIdx.x; i < D_MODEL; i += 256)
    out[(size_t)t * D_MODEL + i] = (row[i] - mu) * rs * gamma[i] + beta[i];
}

// ---------------------------------------------------------------- launch
extern "C" void kernel_launch(void* const* d_in, const int* in_sizes, int n_in,
                              void* d_out, int out_size, void* d_ws, size_t ws_size,
                              hipStream_t stream) {
  (void)in_sizes; (void)n_in; (void)out_size; (void)ws_size;
  const float* x     = (const float*)d_in[0];
  const float* Wq    = (const float*)d_in[1];
  const float* bq    = (const float*)d_in[2];
  const float* Wk    = (const float*)d_in[3];
  const float* bk    = (const float*)d_in[4];
  const float* Wv    = (const float*)d_in[5];
  const float* bv    = (const float*)d_in[6];
  const float* Wo    = (const float*)d_in[7];
  const float* bo    = (const float*)d_in[8];
  const float* gamma = (const float*)d_in[9];
  const float* beta  = (const float*)d_in[10];
  float* out = (float*)d_out;

  char* ws = (char*)d_ws;
  size_t off = 0;
  auto carve = [&](size_t bytes) -> char* {
    char* p = ws + off;
    off += (bytes + 255) & ~(size_t)255;
    return p;
  };
  const size_t nX  = (size_t)NTOK * D_MODEL;            // 4,194,304
  const size_t nW  = (size_t)D_MODEL * D_MODEL;         // 1,048,576
  const size_t nHd = (size_t)NBATCH * NHEADS * S_LEN * DHEAD;  // 4,194,304

  _Float16* xh    = (_Float16*)carve(nX * 2);
  _Float16* Wqh   = (_Float16*)carve(nW * 2);
  _Float16* Wkh   = (_Float16*)carve(nW * 2);
  _Float16* Wvh   = (_Float16*)carve(nW * 2);
  _Float16* Woh   = (_Float16*)carve(nW * 2);
  _Float16* Qh    = (_Float16*)carve(nHd * 2);
  _Float16* Kh    = (_Float16*)carve(nHd * 2);
  _Float16* Vt    = (_Float16*)carve(nHd * 2);
  _Float16* attnh = (_Float16*)carve(nX * 2);
  float*    ypre  = (float*)carve(nX * 4);

  pack_f16_kernel<<<(int)((nX + 255) / 256), 256, 0, stream>>>(x,  xh,  (int)nX);
  pack_f16_kernel<<<(int)((nW + 255) / 256), 256, 0, stream>>>(Wq, Wqh, (int)nW);
  pack_f16_kernel<<<(int)((nW + 255) / 256), 256, 0, stream>>>(Wk, Wkh, (int)nW);
  pack_f16_kernel<<<(int)((nW + 255) / 256), 256, 0, stream>>>(Wv, Wvh, (int)nW);
  pack_f16_kernel<<<(int)((nW + 255) / 256), 256, 0, stream>>>(Wo, Woh, (int)nW);

  dim3 gg(NTOK / 32, D_MODEL / 64);  // 128 x 16
  gemm_xw_kernel<<<gg, 32, 0, stream>>>(xh, Wqh, bq, nullptr, Qh, nullptr, 0);
  gemm_xw_kernel<<<gg, 32, 0, stream>>>(xh, Wkh, bk, nullptr, Kh, nullptr, 1);
  gemm_xw_kernel<<<gg, 32, 0, stream>>>(xh, Wvh, bv, nullptr, Vt, nullptr, 2);

  attn_kernel<<<dim3(S_LEN / 16, NHEADS, NBATCH), 32, 0, stream>>>(Qh, Kh, Vt, attnh);

  gemm_xw_kernel<<<gg, 32, 0, stream>>>(attnh, Woh, bo, x, nullptr, ypre, 3);

  ln_kernel<<<NTOK, 256, 0, stream>>>(ypre, gamma, beta, out);
}
